// QMNN_51599737094576
// MI455X (gfx1250) — matrix operand order repaired and checked by reading, not compiled, not run on hardware
//
#include <hip/hip_runtime.h>
#include <math.h>

// ---------------------------------------------------------------------------
// CDNA5 (gfx1250) implementation. All dense GEMMs use v_wmma_f32_16x16x32_f16
// (probe-verified builtin). Wave32; A/B fragments use contiguous k = 16h + s (same order in both).
// Round-2 changes: 16x64 output strip per wave in the generic GEMM (4 WMMAs
// share one A fragment -> 4x less A traffic), explicit float4 (b128) loads
// for all aligned fragment fetches.
// ---------------------------------------------------------------------------

typedef __attribute__((ext_vector_type(16))) _Float16 v16h;
typedef __attribute__((ext_vector_type(8)))  float    v8f;

#define WMMA(a, b, c) \
  __builtin_amdgcn_wmma_f32_16x16x32_f16(false, (a), false, (b), (short)0, (c), false, false)

// A-matrix 16x32 f16 fragment. Lane L (row = L%16):
//   L<16 : K=0..15
//   L>=16: K=16..31
// Base (row*lda + k0 + {0|16}) is 16B-aligned at every call site.
__device__ __forceinline__ v16h load_a_frag(const float* __restrict__ A, size_t row,
                                            int lda, int k0, int lane) {
  const float* ap = A + row * (size_t)lda + k0 + ((lane & 16) ? 16 : 0);
  float t[16];
  *(float4*)(t + 0)  = *(const float4*)(ap + 0);
  *(float4*)(t + 4)  = *(const float4*)(ap + 4);
  *(float4*)(t + 8)  = *(const float4*)(ap + 8);
  *(float4*)(t + 12) = *(const float4*)(ap + 12);
  v16h a;
#pragma unroll
  for (int j = 0; j < 16; ++j) a[j] = (_Float16)t[j];
  return a;
}

// B-matrix 32x16 f16 fragment, B[k,n] = W[n,k] with W row-major [N,K].
// Lane L: col n = L%16; L<16 holds K=0..15, L>=16 holds K=16..31.
__device__ __forceinline__ v16h load_b_frag_wt(const float* __restrict__ W, size_t ncol,
                                               int ldw, int k0, int lane) {
  const float* wp = W + ncol * (size_t)ldw + k0 + ((lane & 16) ? 16 : 0);
  float t[16];
  *(float4*)(t + 0)  = *(const float4*)(wp + 0);
  *(float4*)(t + 4)  = *(const float4*)(wp + 4);
  *(float4*)(t + 8)  = *(const float4*)(wp + 8);
  *(float4*)(t + 12) = *(const float4*)(wp + 12);
  v16h b;
#pragma unroll
  for (int j = 0; j < 16; ++j) b[j] = (_Float16)t[j];
  return b;
}

// ---------------------------------------------------------------------------
// Generic GEMM:  C[M,N] = act( A[M,K] @ W^T + bias1 (+ bias2) )
// W row-major [N,K]. One 16x64 output strip per wave (4 accumulators share
// one A fragment), 8 waves per block. M%16==0, N%64==0, K%32==0 everywhere.
// ---------------------------------------------------------------------------
__global__ __launch_bounds__(256) void gemm_wmma(const float* __restrict__ A,
                                                 const float* __restrict__ W,
                                                 const float* __restrict__ bias,
                                                 const float* __restrict__ bias2,
                                                 float* __restrict__ C,
                                                 int M, int N, int K, int act) {
  const int lane = threadIdx.x & 31;
  const int wave = threadIdx.x >> 5;
  const int mtiles = M >> 4, n4 = N >> 6;
  const int tile = blockIdx.x * 8 + wave;
  if (tile >= mtiles * n4) return;                    // wave-uniform: EXEC stays full
  const int mt = tile / n4, nt4 = tile % n4;
  const int colb = nt4 * 64 + (lane & 15);
  const size_t arow = (size_t)mt * 16 + (lane & 15);

  v8f acc[4] = {};
  for (int k0 = 0; k0 < K; k0 += 32) {
    v16h a = load_a_frag(A, arow, K, k0, lane);
#pragma unroll
    for (int t = 0; t < 4; ++t) {
      v16h b = load_b_frag_wt(W, (size_t)(colb + t * 16), K, k0, lane);
      acc[t] = WMMA(a, b, acc[t]);
    }
  }
  const int rbase = mt * 16 + ((lane & 16) ? 8 : 0);  // C layout: VGPR r -> row r(+8)
#pragma unroll
  for (int t = 0; t < 4; ++t) {
    const int col = colb + t * 16;
    float bv = bias ? bias[col] : 0.f;
    if (bias2) bv += bias2[col];
#pragma unroll
    for (int r = 0; r < 8; ++r) {
      float v = acc[t][r] + bv;
      if (act == 1) v = fmaxf(v, 0.f);
      C[(size_t)(rbase + r) * N + col] = v;
    }
  }
}

// ---------------------------------------------------------------------------
// Fused sims + argmax:  idx[m] = argmax_n ( Q[m,:] . Keys[n,:] ),  K=256.
// One block per 16 tokens. A-fragments (16x256) preloaded, 8 waves stride
// over the CAP/16 key tiles, per-lane running argmax, shuffle + LDS reduce.
// ---------------------------------------------------------------------------
__global__ __launch_bounds__(256) void sims_argmax(const float* __restrict__ Q,
                                                   const float* __restrict__ Keys,
                                                   int* __restrict__ idx,
                                                   int CAPn, int K) {
  const int lane = threadIdx.x & 31;
  const int wave = threadIdx.x >> 5;
  const size_t arow = (size_t)blockIdx.x * 16 + (lane & 15);

  v16h af[8];
#pragma unroll
  for (int kk = 0; kk < 8; ++kk) af[kk] = load_a_frag(Q, arow, K, kk * 32, lane);

  float best[8];
  int   bidx[8];
#pragma unroll
  for (int r = 0; r < 8; ++r) { best[r] = -1e30f; bidx[r] = 0; }

  const int ntiles = CAPn >> 4;
  for (int nt = wave; nt < ntiles; nt += 8) {
    const int col = nt * 16 + (lane & 15);
    v8f acc = {};
#pragma unroll
    for (int kk = 0; kk < 8; ++kk) {
      v16h b = load_b_frag_wt(Keys, (size_t)col, K, kk * 32, lane);
      acc = WMMA(af[kk], b, acc);
    }
#pragma unroll
    for (int r = 0; r < 8; ++r) {
      float v = acc[r];
      if (v > best[r] || (v == best[r] && col < bidx[r])) { best[r] = v; bidx[r] = col; }
    }
  }
  // reduce across the 16 lanes of each half (xor 1,2,4,8 stays in-half)
#pragma unroll
  for (int off = 1; off < 16; off <<= 1) {
#pragma unroll
    for (int r = 0; r < 8; ++r) {
      float ov = __shfl_xor(best[r], off, 32);
      int   oi = __shfl_xor(bidx[r], off, 32);
      if (ov > best[r] || (ov == best[r] && oi < bidx[r])) { best[r] = ov; bidx[r] = oi; }
    }
  }
  __shared__ float sbest[8][16];
  __shared__ int   sidx[8][16];
  if ((lane & 15) == 0) {
    const int rb = (lane & 16) ? 8 : 0;
#pragma unroll
    for (int r = 0; r < 8; ++r) { sbest[wave][rb + r] = best[r]; sidx[wave][rb + r] = bidx[r]; }
  }
  __syncthreads();
  if (threadIdx.x < 16) {
    const int row = threadIdx.x;
    float bv = sbest[0][row]; int bi = sidx[0][row];
#pragma unroll
    for (int w = 1; w < 8; ++w) {
      float v = sbest[w][row]; int i2 = sidx[w][row];
      if (v > bv || (v == bv && i2 < bi)) { bv = v; bi = i2; }
    }
    idx[blockIdx.x * 16 + row] = bi;
  }
}

// ---------------------------------------------------------------------------
// LSTM recurrence (x-projection precomputed as a GEMM). One block per batch
// row; h/c/g live in LDS; thread n owns gate column n of 4H=1024.
// ---------------------------------------------------------------------------
__device__ __forceinline__ float sigmoidf_(float x) { return 1.f / (1.f + __expf(-x)); }

__global__ __launch_bounds__(1024) void lstm_scan(const float* __restrict__ xp,   // [B,S,1024]
                                                  const float* __restrict__ whh,  // [1024,256]
                                                  float* __restrict__ hout,       // [B,S,256]
                                                  int S) {
  __shared__ float h[256];
  __shared__ float c[256];
  __shared__ float g[1024];
  const int b = blockIdx.x;
  const int n = threadIdx.x;
  if (n < 256) { h[n] = 0.f; c[n] = 0.f; }
  const float* w = whh + (size_t)n * 256;
  for (int t = 0; t < S; ++t) {
    __syncthreads();
    float acc = xp[(((size_t)b * S) + t) * 1024 + n];
#pragma unroll 4
    for (int k = 0; k < 256; ++k) acc += h[k] * w[k];
    g[n] = acc;
    __syncthreads();
    if (n < 256) {
      float ig = sigmoidf_(g[n]);
      float fg = sigmoidf_(g[256 + n]);
      float gg = tanhf(g[512 + n]);
      float og = sigmoidf_(g[768 + n]);
      float cn = fg * c[n] + ig * gg;
      c[n] = cn;
      float hn = og * tanhf(cn);
      h[n] = hn;
      hout[(((size_t)b * S) + t) * 256 + n] = hn;
    }
  }
}

// ---------------------------------------------------------------------------
// Memory gather + both quantum layers fused. One block per token.
// ---------------------------------------------------------------------------
__global__ __launch_bounds__(256) void quantum2(const int* __restrict__ idx,
                                                const float* __restrict__ memv,
                                                const float* __restrict__ w1a, const float* __restrict__ b1a,
                                                const float* __restrict__ qpa, const float* __restrict__ w2a,
                                                const float* __restrict__ b2a,
                                                const float* __restrict__ w1b, const float* __restrict__ b1b,
                                                const float* __restrict__ qpb, const float* __restrict__ w2b,
                                                const float* __restrict__ b2b,
                                                float* __restrict__ outp) {
  __shared__ float buf[256];
  __shared__ float st[8];
  const int tok = blockIdx.x;
  const int t = threadIdx.x;
  const float HPI = 1.57079632679f;
  const int m = idx[tok];
  buf[t] = memv[(size_t)m * 256 + t];
  __syncthreads();
  // layer A
  if (t < 8) {
    float acc = b1a[t];
    const float* w = w1a + t * 256;
#pragma unroll 4
    for (int k = 0; k < 256; ++k) acc += buf[k] * w[k];
    float z = tanhf(acc);
    st[t] = __cosf(z * HPI) * __cosf(qpa[t * 3 + 0]) + __sinf(qpa[t * 3 + 1]) * __cosf(qpa[t * 3 + 2]);
  }
  __syncthreads();
  float o1 = b2a[t];
#pragma unroll
  for (int q = 0; q < 8; ++q) o1 += st[q] * w2a[t * 8 + q];
  __syncthreads();
  buf[t] = o1;
  __syncthreads();
  // layer B
  if (t < 8) {
    float acc = b1b[t];
    const float* w = w1b + t * 256;
#pragma unroll 4
    for (int k = 0; k < 256; ++k) acc += buf[k] * w[k];
    float z = tanhf(acc);
    st[t] = __cosf(z * HPI) * __cosf(qpb[t * 3 + 0]) + __sinf(qpb[t * 3 + 1]) * __cosf(qpb[t * 3 + 2]);
  }
  __syncthreads();
  float o2 = b2b[t];
#pragma unroll
  for (int q = 0; q < 8; ++q) o2 += st[q] * w2b[t * 8 + q];
  outp[(size_t)tok * 256 + t] = o2;
}

// ---------------------------------------------------------------------------
// Flash attention: NH=4, hd=64, scale=1/8. Block = 4 waves = 64 q rows;
// each wave owns 16 q rows, streams keys in 32-tiles: 4 WMMAs (QK^T, K=64)
// + 4 WMMAs (P@V over hd=64). Online softmax with shuffle reductions.
// ---------------------------------------------------------------------------
__global__ __launch_bounds__(128) void flash_attn_kernel(const float* __restrict__ Qh,
                                                         const float* __restrict__ Kh,
                                                         const float* __restrict__ Vh,
                                                         float* __restrict__ Oatt, int S) {
  const int E = 256, HD = 64;
  const int lane = threadIdx.x & 31;
  const int wave = threadIdx.x >> 5;
  const int b = blockIdx.z, h = blockIdx.y;
  const int qbase = blockIdx.x * 64 + wave * 16;
  const float* Qp = Qh + ((size_t)b * S) * E + h * HD;
  const float* Kp = Kh + ((size_t)b * S) * E + h * HD;
  const float* Vp = Vh + ((size_t)b * S) * E + h * HD;
  const int c0 = lane & 15;
  const int rb = (lane & 16) ? 8 : 0;
  const int half = (lane & 16) ? 8 : 0;

  // Q fragments: A 16x64 as two 16x32 chunks, kept in registers all iteration.
  v16h qf[2];
#pragma unroll
  for (int c = 0; c < 2; ++c)
    qf[c] = load_a_frag(Qp, (size_t)(qbase + c0), E, c * 32, lane);

  v8f o[4] = {};
  float mrow[8], lrow[8];
#pragma unroll
  for (int r = 0; r < 8; ++r) { mrow[r] = -1e30f; lrow[r] = 0.f; }

  __shared__ _Float16 plds[4][16][32];   // wave-private P staging (C-layout -> A-layout)
  const float scale = 0.125f;            // 1/sqrt(64)

  for (int k0 = 0; k0 < S; k0 += 32) {
    v8f s0 = {}, s1 = {};
#pragma unroll
    for (int c = 0; c < 2; ++c) {
      // B = K^T fragments: 16 consecutive dims per lane -> b128 loads
      const int kb = c * 32 + ((lane & 16) ? 16 : 0);
      const float* kp0 = Kp + (size_t)(k0 + c0) * E + kb;
      const float* kp1 = Kp + (size_t)(k0 + 16 + c0) * E + kb;
      float t0[16], t1[16];
#pragma unroll
      for (int q = 0; q < 4; ++q) {
        *(float4*)(t0 + 4 * q) = *(const float4*)(kp0 + 4 * q);
        *(float4*)(t1 + 4 * q) = *(const float4*)(kp1 + 4 * q);
      }
      v16h bf0, bf1;
#pragma unroll
      for (int j = 0; j < 16; ++j) { bf0[j] = (_Float16)t0[j]; bf1[j] = (_Float16)t1[j]; }
      s0 = WMMA(qf[c], bf0, s0);
      s1 = WMMA(qf[c], bf1, s1);
    }
    // online softmax across this 16x32 score slab
    float p0[8], p1[8], corr[8];
#pragma unroll
    for (int r = 0; r < 8; ++r) {
      float a0 = s0[r] * scale;
      float a1 = s1[r] * scale;
      float cm = fmaxf(a0, a1);
      cm = fmaxf(cm, __shfl_xor(cm, 1, 32));
      cm = fmaxf(cm, __shfl_xor(cm, 2, 32));
      cm = fmaxf(cm, __shfl_xor(cm, 4, 32));
      cm = fmaxf(cm, __shfl_xor(cm, 8, 32));
      float mn = fmaxf(mrow[r], cm);
      float e0 = __expf(a0 - mn);
      float e1 = __expf(a1 - mn);
      float rs = e0 + e1;
      rs += __shfl_xor(rs, 1, 32);
      rs += __shfl_xor(rs, 2, 32);
      rs += __shfl_xor(rs, 4, 32);
      rs += __shfl_xor(rs, 8, 32);
      float cr = __expf(mrow[r] - mn);
      lrow[r] = lrow[r] * cr + rs;
      mrow[r] = mn;
      p0[r] = e0; p1[r] = e1; corr[r] = cr;
    }
#pragma unroll
    for (int d = 0; d < 4; ++d)
#pragma unroll
      for (int r = 0; r < 8; ++r) o[d][r] *= corr[r];
    // stage P (C layout) to LDS, read back as A fragment (wave-internal, DS in-order)
#pragma unroll
    for (int r = 0; r < 8; ++r) {
      plds[wave][rb + r][c0]      = (_Float16)p0[r];
      plds[wave][rb + r][16 + c0] = (_Float16)p1[r];
    }
    v16h pf;
#pragma unroll
    for (int j = 0; j < 8; ++j) pf[j] = plds[wave][c0][2 * half + j];
#pragma unroll
    for (int j = 8; j < 16; ++j) pf[j] = plds[wave][c0][2 * half + j];
    // P @ V over the 4 hd-subtiles (V column access: strided scalar loads)
    const int kk = (lane & 16) ? 16 : 0;
#pragma unroll
    for (int d = 0; d < 4; ++d) {
      const float* vp = Vp + (size_t)(k0 + kk) * E + d * 16 + c0;
      v16h bf;
#pragma unroll
      for (int j = 0; j < 16; ++j) bf[j] = (_Float16)vp[(size_t)j * E];
      o[d] = WMMA(pf, bf, o[d]);
    }
  }
#pragma unroll
  for (int r = 0; r < 8; ++r) {
    const float inv = 1.f / lrow[r];
    const size_t rowg = (size_t)(b * S + qbase + rb + r) * E + h * HD;
#pragma unroll
    for (int d = 0; d < 4; ++d) Oatt[rowg + d * 16 + c0] = o[d][r] * inv;
  }
}

// ---------------------------------------------------------------------------
__global__ void concat2(const float* __restrict__ a, const float* __restrict__ b,
                        float* __restrict__ c, int M) {
  int i = blockIdx.x * blockDim.x + threadIdx.x;
  if (i >= M * 512) return;
  int m = i >> 9, k = i & 511;
  c[i] = (k < 256) ? a[(size_t)m * 256 + k] : b[(size_t)m * 256 + (k - 256)];
}

// ---------------------------------------------------------------------------
static inline int gemm_blocks(int M, int N) { return ((M >> 4) * (N >> 6) + 7) / 8; }

extern "C" void kernel_launch(void* const* d_in, const int* in_sizes, int n_in,
                              void* d_out, int out_size, void* d_ws, size_t ws_size,
                              hipStream_t stream) {
  (void)in_sizes; (void)n_in; (void)out_size; (void)ws_size;
  const float* x     = (const float*)d_in[0];
  const float* ew1   = (const float*)d_in[1];
  const float* eb1   = (const float*)d_in[2];
  const float* ew2   = (const float*)d_in[3];
  const float* eb2   = (const float*)d_in[4];
  const float* wih0  = (const float*)d_in[5];
  const float* whh0  = (const float*)d_in[6];
  const float* bih0  = (const float*)d_in[7];
  const float* bhh0  = (const float*)d_in[8];
  const float* wih1  = (const float*)d_in[9];
  const float* whh1  = (const float*)d_in[10];
  const float* bih1  = (const float*)d_in[11];
  const float* bhh1  = (const float*)d_in[12];
  const float* mkeys = (const float*)d_in[13];
  const float* mvals = (const float*)d_in[14];
  const float* q0w1  = (const float*)d_in[15];
  const float* q0b1  = (const float*)d_in[16];
  const float* q0qp  = (const float*)d_in[17];
  const float* q0w2  = (const float*)d_in[18];
  const float* q0b2  = (const float*)d_in[19];
  const float* q1w1  = (const float*)d_in[20];
  const float* q1b1  = (const float*)d_in[21];
  const float* q1qp  = (const float*)d_in[22];
  const float* q1w2  = (const float*)d_in[23];
  const float* q1b2  = (const float*)d_in[24];
  const float* wqkv  = (const float*)d_in[25];
  const float* bqkv  = (const float*)d_in[26];
  const float* wo    = (const float*)d_in[27];
  const float* bo    = (const float*)d_in[28];
  const float* dw1   = (const float*)d_in[29];
  const float* db1   = (const float*)d_in[30];
  const float* dw2   = (const float*)d_in[31];
  const float* db2   = (const float*)d_in[32];

  const int M = 8 * 2048;   // 16384 tokens
  const int S = 2048;

  float* ws   = (float*)d_ws;
  float* bufA = ws;                   // 16,777,216 floats: enc1 / xp0 / xp1 / qkv / combined
  float* enc  = ws + 16777216;        // 4,194,304: enc, then qp
  float* h0   = ws + 20971520;        // 4,194,304: h0, then att
  float* ctrl = ws + 25165824;        // 4,194,304
  float* dec1 = ws + 29360128;        // 4,194,304
  int*   idx  = (int*)(ws + 33554432);
  float* qh = bufA, *kh = bufA + 4194304, *vh = bufA + 8388608;
  float* combined = bufA;
  float* att = h0;
  float* qp  = enc;

  float* outp     = (float*)d_out;          // [16384, 512]
  float* attended = outp + 8388608;         // [16384, 256]

  // encoder
  gemm_wmma<<<gemm_blocks(M, 256), 256, 0, stream>>>(x, ew1, eb1, nullptr, bufA, M, 256, 512, 1);
  gemm_wmma<<<gemm_blocks(M, 256), 256, 0, stream>>>(bufA, ew2, eb2, nullptr, enc, M, 256, 256, 0);
  // LSTM layer 0: batched x-projection, then sequential scan
  gemm_wmma<<<gemm_blocks(M, 1024), 256, 0, stream>>>(enc, wih0, bih0, bhh0, bufA, M, 1024, 256, 0);
  lstm_scan<<<8, 1024, 0, stream>>>(bufA, whh0, h0, S);
  // LSTM layer 1
  gemm_wmma<<<gemm_blocks(M, 1024), 256, 0, stream>>>(h0, wih1, bih1, bhh1, bufA, M, 1024, 256, 0);
  lstm_scan<<<8, 1024, 0, stream>>>(bufA, whh1, ctrl, S);
  // memory lookup (fused sims + argmax), then gather + quantum layers
  sims_argmax<<<M / 16, 256, 0, stream>>>(enc, mkeys, idx, 8192, 256);
  quantum2<<<M, 256, 0, stream>>>(idx, mvals, q0w1, q0b1, q0qp, q0w2, q0b2,
                                  q1w1, q1b1, q1qp, q1w2, q1b2, qp);
  // attention: projections, flash core, output projection
  gemm_wmma<<<gemm_blocks(M, 256), 256, 0, stream>>>(ctrl, wqkv,           bqkv,       nullptr, qh, M, 256, 256, 0);
  gemm_wmma<<<gemm_blocks(M, 256), 256, 0, stream>>>(qp,   wqkv + 65536,   bqkv + 256, nullptr, kh, M, 256, 256, 0);
  gemm_wmma<<<gemm_blocks(M, 256), 256, 0, stream>>>(qp,   wqkv + 131072,  bqkv + 512, nullptr, vh, M, 256, 256, 0);
  dim3 fg(S / 64, 4, 8);
  flash_attn_kernel<<<fg, 128, 0, stream>>>(qh, kh, vh, att, S);
  gemm_wmma<<<gemm_blocks(M, 256), 256, 0, stream>>>(att, wo, bo, nullptr, attended, M, 256, 256, 0);
  // decoder on concat(ctrl, attended)
  concat2<<<(M * 512 + 255) / 256, 256, 0, stream>>>(ctrl, attended, combined, M);
  gemm_wmma<<<gemm_blocks(M, 256), 256, 0, stream>>>(combined, dw1, db1, nullptr, dec1, M, 256, 512, 1);
  gemm_wmma<<<gemm_blocks(M, 512), 256, 0, stream>>>(dec1, dw2, db2, nullptr, outp, M, 512, 256, 0);
}
